// GraphAutoencoder_73203422593436
// MI455X (gfx1250) — matrix operand order repaired and checked by reading, not compiled, not run on hardware
//
#include <hip/hip_runtime.h>
#include <hip/hip_bf16.h>

typedef __attribute__((ext_vector_type(16))) __bf16 v16bf;
typedef __attribute__((ext_vector_type(8)))  float  v8f;

#define GN    100000
#define GE    1600000
#define IN_C  256
#define HID_C 128
#define OUT_C 256

// ---------------------------------------------------------------------------
// fp32 -> bf16 conversion (plain)
// ---------------------------------------------------------------------------
__global__ void cvt_f32_to_bf16(const float* __restrict__ in,
                                __hip_bfloat16* __restrict__ out, int n) {
    int i = blockIdx.x * blockDim.x + threadIdx.x;
    if (i < n) out[i] = __float2bfloat16(in[i]);
}

// fp32 KxM -> bf16 MxK (transposed, so GEMM B-fragments are contiguous)
__global__ void cvt_w_transpose(const float* __restrict__ in,
                                __hip_bfloat16* __restrict__ out,
                                int Kdim, int Mdim) {
    int i = blockIdx.x * blockDim.x + threadIdx.x;
    if (i < Kdim * Mdim) {
        int k = i / Mdim, m = i - k * Mdim;
        out[(size_t)m * Kdim + k] = __float2bfloat16(in[i]);
    }
}

// ---------------------------------------------------------------------------
// degree / normalization
// ---------------------------------------------------------------------------
__global__ void deg_init(float* __restrict__ deg, int n) {
    int i = blockIdx.x * blockDim.x + threadIdx.x;
    if (i < n) deg[i] = 1.0f;                       // self-loop contribution
}

__global__ void deg_count(const int* __restrict__ dst, float* __restrict__ deg, int e) {
    int i = blockIdx.x * blockDim.x + threadIdx.x;
    if (i < e) atomicAdd(&deg[dst[i]], 1.0f);
}

__global__ void deg_rsqrt(const float* __restrict__ deg, float* __restrict__ dinv, int n) {
    int i = blockIdx.x * blockDim.x + threadIdx.x;
    if (i < n) dinv[i] = rsqrtf(deg[i]);            // deg >= 1 always
}

// ---------------------------------------------------------------------------
// GEMM: out[row][col] = (A[row][:] . W[:][col]) * dinv[row]
//   A : nrows x K   bf16 row-major
//   Bt: MOUT  x K   bf16 (W pre-transposed -> contiguous K per column)
// One wave computes a 16 x (16*CT) output slab: A fragment loaded once per
// K-step and reused across CT column tiles (CT back-to-back WMMAs).
// ---------------------------------------------------------------------------
template<int K, int MOUT, int CT>
__global__ void gemm_wmma_scale(const __bf16* __restrict__ A,
                                const __bf16* __restrict__ Bt,
                                const float*  __restrict__ dinv,
                                float*        __restrict__ out,
                                int nrows) {
    const int wave = threadIdx.x >> 5;
    const int lane = threadIdx.x & 31;
    const int row_tile = blockIdx.x * (blockDim.x >> 5) + wave;
    if (row_tile * 16 >= nrows) return;             // wave-uniform: EXEC stays all-1s
    const int colg0 = blockIdx.y * (CT * 16);

    const int m  = lane & 15;                       // A: row-in-tile / B,C,D: col-in-tile
    const int kg = lane >> 4;                       // lane half selector
    const int r0 = row_tile * 16;

    const __bf16* arow = A + (size_t)(r0 + m) * K;
    v8f acc[CT];
    #pragma unroll
    for (int ct = 0; ct < CT; ++ct) acc[ct] = (v8f){};

    for (int k0 = 0; k0 < K; k0 += 32) {
        // A fragment (16x32 bf16): lane holds K = [k0+16kg .. +15]
        union { v16bf v; uint4 u[2]; } af;
        af.u[0] = *(const uint4*)(arow + k0 + kg * 16);
        af.u[1] = *(const uint4*)(arow + k0 + kg * 16 + 8);

        #pragma unroll
        for (int ct = 0; ct < CT; ++ct) {
            // B fragment (32x16 bf16): lane holds col = colg0+16ct+m,
            // K = k0+16kg .. +15  -> 32 contiguous bytes in transposed W
            union { v16bf v; uint4 u[2]; } bf;
            const __bf16* bp = Bt + (size_t)(colg0 + ct * 16 + m) * K + k0 + kg * 16;
            bf.u[0] = *(const uint4*)(bp);
            bf.u[1] = *(const uint4*)(bp + 8);

            acc[ct] = __builtin_amdgcn_wmma_f32_16x16x32_bf16(
                          false, af.v, false, bf.v, (short)0, acc[ct], false, false);
        }
    }

    // C/D layout: VGPR r, lane -> (M = r + 8*kg, N = m)
    #pragma unroll
    for (int r = 0; r < 8; ++r) {
        const int row = r0 + kg * 8 + r;
        const float s = dinv[row];
        float* orow = out + (size_t)row * MOUT + colg0 + m;
        #pragma unroll
        for (int ct = 0; ct < CT; ++ct)
            orow[ct * 16] = acc[ct][r] * s;
    }
}

// ---------------------------------------------------------------------------
// edge scatter-add: one wave per edge, lanes stride over channels
//   acc[dst][c] += g[src][c]   (L2-resident fp32 atomics)
// ---------------------------------------------------------------------------
template<int C>
__global__ void edge_scatter(const int* __restrict__ src,
                             const int* __restrict__ dst,
                             const float* __restrict__ g,
                             float* __restrict__ acc, int e) {
    const int lane = threadIdx.x & 31;
    const int ei   = blockIdx.x * (blockDim.x >> 5) + (threadIdx.x >> 5);
    if (ei >= e) return;
    const int s = src[ei];
    const int d = dst[ei];
    const float* gs = g   + (size_t)s * C;
    float*       ad = acc + (size_t)d * C;
    #pragma unroll
    for (int c = lane; c < C; c += 32)
        atomicAdd(ad + c, gs[c]);
}

// ---------------------------------------------------------------------------
// epilogues: h = relu(dinv[row]*acc + bias) -> bf16 ; out = dinv[row]*acc + bias
// ---------------------------------------------------------------------------
template<int C>
__global__ void epilogue_relu_bf16(const float* __restrict__ acc,
                                   const float* __restrict__ dinv,
                                   const float* __restrict__ bias,
                                   __hip_bfloat16* __restrict__ out, int n) {
    int i = blockIdx.x * blockDim.x + threadIdx.x;
    if (i < n) {
        int row = i / C, c = i - row * C;
        float v = acc[i] * dinv[row] + bias[c];
        out[i] = __float2bfloat16(v > 0.0f ? v : 0.0f);
    }
}

template<int C>
__global__ void epilogue_f32(const float* __restrict__ acc,
                             const float* __restrict__ dinv,
                             const float* __restrict__ bias,
                             float* __restrict__ out, int n) {
    int i = blockIdx.x * blockDim.x + threadIdx.x;
    if (i < n) {
        int row = i / C, c = i - row * C;
        out[i] = acc[i] * dinv[row] + bias[c];
    }
}

// ---------------------------------------------------------------------------
extern "C" void kernel_launch(void* const* d_in, const int* in_sizes, int n_in,
                              void* d_out, int out_size, void* d_ws, size_t ws_size,
                              hipStream_t stream) {
    const float* x   = (const float*)d_in[0];
    const int*   ei  = (const int*)  d_in[1];
    const float* W1  = (const float*)d_in[2];
    const float* b1  = (const float*)d_in[3];
    const float* W2  = (const float*)d_in[4];
    const float* b2  = (const float*)d_in[5];
    const int* src = ei;
    const int* dst = ei + GE;

    char* ws = (char*)d_ws;
    // workspace layout (~209 MB total, with region reuse across the pipeline)
    const size_t o_deg  = 0;                              // N f32
    const size_t o_dinv = 1u << 20;                       // N f32
    const size_t o_w1b  = 2u << 20;                       // 128x256 bf16 (transposed)
    const size_t o_w2b  = o_w1b + (size_t)IN_C * HID_C * 2;
    const size_t o_A    = 4u << 20;                       // region A: 102.4 MB
    const size_t o_xb   = o_A;                            // x bf16 (51.2 MB)
    const size_t o_g1   = o_A + (size_t)GN * IN_C * 2;    // g1 f32 (51.2 MB)
    const size_t o_g2   = o_A;                            // g2 f32 (102.4 MB, reuses xb+g1)
    const size_t o_B    = o_A + (size_t)GN * OUT_C * 4;   // region B: 102.4 MB
    const size_t o_acc1 = o_B;                            // acc1 f32 (51.2 MB)
    const size_t o_h1b  = o_B + (size_t)GN * HID_C * 4;   // h1 bf16 (25.6 MB)
    const size_t o_acc2 = o_B;                            // acc2 f32 (102.4 MB, reuses acc1+h1b)

    float*          deg  = (float*)(ws + o_deg);
    float*          dinv = (float*)(ws + o_dinv);
    __hip_bfloat16* w1b  = (__hip_bfloat16*)(ws + o_w1b);
    __hip_bfloat16* w2b  = (__hip_bfloat16*)(ws + o_w2b);
    __hip_bfloat16* xb   = (__hip_bfloat16*)(ws + o_xb);
    float*          g1   = (float*)(ws + o_g1);
    float*          g2   = (float*)(ws + o_g2);
    float*          acc1 = (float*)(ws + o_acc1);
    __hip_bfloat16* h1b  = (__hip_bfloat16*)(ws + o_h1b);
    float*          acc2 = (float*)(ws + o_acc2);
    float*          out  = (float*)d_out;

    const int T = 256;
    // --- precision conversion (weights transposed for contiguous B loads) ----
    {   int n = GN * IN_C;
        cvt_f32_to_bf16<<<(n + T - 1) / T, T, 0, stream>>>(x, xb, n); }
    {   int n = IN_C * HID_C;
        cvt_w_transpose<<<(n + T - 1) / T, T, 0, stream>>>(W1, w1b, IN_C, HID_C); }
    {   int n = HID_C * OUT_C;
        cvt_w_transpose<<<(n + T - 1) / T, T, 0, stream>>>(W2, w2b, HID_C, OUT_C); }

    // --- degree + dinv --------------------------------------------------------
    deg_init <<<(GN + T - 1) / T, T, 0, stream>>>(deg, GN);
    deg_count<<<(GE + T - 1) / T, T, 0, stream>>>(dst, deg, GE);
    deg_rsqrt<<<(GN + T - 1) / T, T, 0, stream>>>(deg, dinv, GN);

    const int row_tiles = GN / 16;                     // 6250 exactly
    const int gemm_gx   = (row_tiles + 7) / 8;         // 8 waves/block

    // --- layer 1 --------------------------------------------------------------
    gemm_wmma_scale<IN_C, HID_C, 8>
        <<<dim3(gemm_gx, HID_C / 128), T, 0, stream>>>(
            (const __bf16*)xb, (const __bf16*)w1b, dinv, g1, GN);

    hipMemcpyAsync(acc1, g1, (size_t)GN * HID_C * 4,
                   hipMemcpyDeviceToDevice, stream);   // self-loop term

    edge_scatter<HID_C><<<(GE + 7) / 8, T, 0, stream>>>(src, dst, g1, acc1, GE);

    {   int n = GN * HID_C;
        epilogue_relu_bf16<HID_C><<<(n + T - 1) / T, T, 0, stream>>>(
            acc1, dinv, b1, h1b, n); }

    // --- layer 2 --------------------------------------------------------------
    gemm_wmma_scale<HID_C, OUT_C, 8>
        <<<dim3(gemm_gx, OUT_C / 128), T, 0, stream>>>(
            (const __bf16*)h1b, (const __bf16*)w2b, dinv, g2, GN);

    hipMemcpyAsync(acc2, g2, (size_t)GN * OUT_C * 4,
                   hipMemcpyDeviceToDevice, stream);

    edge_scatter<OUT_C><<<(GE + 7) / 8, T, 0, stream>>>(src, dst, g2, acc2, GE);

    {   int n = GN * OUT_C;
        epilogue_f32<OUT_C><<<(n + T - 1) / T, T, 0, stream>>>(
            acc2, dinv, b2, out, n); }
}
